// GQAttention_2817498546271
// MI455X (gfx1250) — matrix operand order repaired and checked by reading, not compiled, not run on hardware
//
#include <hip/hip_runtime.h>
#include <hip/hip_bf16.h>
#include <cstdint>

// ---------------------------------------------------------------------------
// GQA attention forward for MI455X (gfx1250, wave32, WMMA bf16->f32, TDM).
// ---------------------------------------------------------------------------

typedef __bf16 bf16_t;
typedef __attribute__((ext_vector_type(16))) __bf16 v16bf;
typedef __attribute__((ext_vector_type(8)))  __bf16 v8bf;
typedef __attribute__((ext_vector_type(8)))  float  v8f;
typedef __attribute__((ext_vector_type(4)))  unsigned int u32x4;
typedef __attribute__((ext_vector_type(8)))  unsigned int u32x8;

static constexpr int B_   = 2;
static constexpr int T_   = 2048;
static constexpr int D_   = 2048;
static constexpr int H_   = 16;
static constexpr int HKV_ = 4;
static constexpr int REP_ = H_ / HKV_;   // 4
static constexpr int DH_  = D_ / H_;     // 128
static constexpr int RD_  = 64;

// ---------------------------------------------------------------------------
// fp32 -> bf16 cast
// ---------------------------------------------------------------------------
__global__ void cast_f32_bf16(const float* __restrict__ in,
                              bf16_t* __restrict__ out, int n) {
  int i = blockIdx.x * blockDim.x + threadIdx.x;
  if (i < n) out[i] = (bf16_t)in[i];
}

// ---------------------------------------------------------------------------
// GEMM:  out[n, m] = sum_k A[n, k] * W[m, k]   (A:[N,K] rm, W:[M,K] rm)
// One wave per block; block computes a 16 x 128 output tile.
// ---------------------------------------------------------------------------
template <bool OUT_F32>
__global__ void __launch_bounds__(32)
gemm_bf16_kernel(const bf16_t* __restrict__ A,
                 const bf16_t* __restrict__ W,
                 float* __restrict__ outF,
                 bf16_t* __restrict__ outB,
                 int N, int M, int K) {
  const int lane = threadIdx.x & 31;
  const int half = lane >> 4;
  const int l15  = lane & 15;
  const int nrow = blockIdx.x * 16;
  const int mcol = blockIdx.y * 128;

  v8f acc[8] = {};

  const bf16_t* arow = A + (size_t)(nrow + l15) * K;

  for (int k0 = 0; k0 < K; k0 += 32) {
    if (k0 + 64 < K)                              // stream next A slab toward WGP
      __builtin_prefetch(arow + k0 + 64, 0, 1);   // -> global_prefetch_b8

    v16bf a;
    {
      const bf16_t* p  = arow + k0 + half * 16;
      v8bf lo = *(const v8bf*)(p);
      v8bf hi = *(const v8bf*)(p + 8);
#pragma unroll
      for (int i = 0; i < 8; ++i) { a[i] = lo[i]; a[8 + i] = hi[i]; }
    }
#pragma unroll
    for (int f = 0; f < 8; ++f) {
      const bf16_t* q = W + (size_t)(mcol + f * 16 + l15) * K + k0 + half * 16;
      v16bf b = *(const v16bf*)q;
      acc[f] = __builtin_amdgcn_wmma_f32_16x16x32_bf16(
          false, a, false, b, (short)0, acc[f], false, false);
    }
  }

#pragma unroll
  for (int f = 0; f < 8; ++f) {
    int col = mcol + f * 16 + l15;
#pragma unroll
    for (int r = 0; r < 8; ++r) {
      int row = nrow + r + 8 * half;
      if (OUT_F32) outF[(size_t)row * M + col] = acc[f][r];
      else         outB[(size_t)row * M + col] = (bf16_t)acc[f][r];
    }
  }
}

// ---------------------------------------------------------------------------
// RoPE (+ optional per-head temperature), in place on bf16 [B*T, NH*DH]
// ---------------------------------------------------------------------------
__global__ void rope_kernel(bf16_t* __restrict__ qk,
                            const float* __restrict__ cosT,
                            const float* __restrict__ sinT,
                            const float* __restrict__ temp,
                            int NH, int applyTemp) {
  const int row = blockIdx.x;        // b*T + t
  const int t   = row % T_;
  const int h   = blockIdx.y;
  const int d   = threadIdx.x;       // 0..DH-1
  bf16_t* p = qk + (size_t)row * ((size_t)NH * DH_) + (size_t)h * DH_;
  const float scale = applyTemp ? temp[h] : 1.0f;

  float v;
  if (d < RD_) {
    const int i = d >> 1;
    const float c  = cosT[t * (RD_ / 2) + i];
    const float s  = sinT[t * (RD_ / 2) + i];
    const float x1 = (float)p[2 * i];
    const float x2 = (float)p[2 * i + 1];
    v = (d & 1) ? (x1 * s + x2 * c) : (x1 * c - x2 * s);
  } else {
    v = (float)p[d];
  }
  __syncthreads();                   // all pair reads before any writes
  p[d] = (bf16_t)(v * scale);
}

// ---------------------------------------------------------------------------
// TDM: async DMA of a 32(keys) x 128(dh) bf16 tile, global -> LDS (row-major).
// 2-D descriptor: tile_dim0=128 contiguous elems, tile_dim1=32 lines,
// line stride 512 elems (HKV*DH). Tracked by TENSORcnt.
// ---------------------------------------------------------------------------
__device__ __forceinline__ void tdm_load_v_tile(const bf16_t* gptr,
                                                unsigned lds_off) {
  const unsigned long long ga = (unsigned long long)(uintptr_t)gptr;
  u32x4 g0;
  g0[0] = 1u;                                            // count=1 (user D#)
  g0[1] = lds_off;                                       // lds_addr
  g0[2] = (unsigned)(ga & 0xffffffffu);                  // global_addr[31:0]
  g0[3] = (unsigned)((ga >> 32) & 0x01ffffffu)           // global_addr[56:32]
          | (2u << 30);                                  // type=2 ("image")
  u32x8 g1;
  g1[0] = (1u << 16);                                    // data_size=1 (2 bytes)
  g1[1] = ((unsigned)DH_ << 16);                         // tensor_dim0[15:0]=128
  g1[2] = (((unsigned)T_ & 0xffffu) << 16);              // tensor_dim1[15:0]=2048
  g1[3] = ((unsigned)DH_ << 16);                         // tile_dim0=128
  g1[4] = 32u;                                           // tile_dim1=32, tile_dim2=0
  g1[5] = (unsigned)(HKV_ * DH_);                        // tensor_dim0_stride=512
  g1[6] = 0u;
  g1[7] = 0u;
  asm volatile("tensor_load_to_lds %0, %1" :: "s"(g0), "s"(g1) : "memory");
}

// ---------------------------------------------------------------------------
// Flash-style causal GQA attention. One wave per block = (b, h, 16 q rows).
// Keys streamed 32/iter: V tile double-buffered via TDM, S = Q K^T (8 wmma),
// online softmax, P relayout through LDS, O += P V (8 wmma).
// ---------------------------------------------------------------------------
__global__ void __launch_bounds__(32)
attn_kernel(const bf16_t* __restrict__ Q,   // [B*T, H*DH]
            const bf16_t* __restrict__ Km,  // [B*T, HKV*DH]
            const bf16_t* __restrict__ Vm,  // [B*T, HKV*DH]
            bf16_t* __restrict__ O) {       // [B*T, H*DH]
  __shared__ __align__(32) bf16_t lds_v[2][32][DH_];  // V tiles, row-major (16 KB)
  __shared__ __align__(32) bf16_t lds_p[16][32];      // P tile (1 KB)

  const int lane = threadIdx.x & 31;
  const int half = lane >> 4;
  const int l15  = lane & 15;

  const int qt = blockIdx.x;          // q tile index
  const int h  = blockIdx.y;
  const int b  = blockIdx.z;
  const int hk = h / REP_;
  const int q0 = qt * 16;

  const size_t qld  = (size_t)H_ * DH_;
  const size_t kvld = (size_t)HKV_ * DH_;
  const bf16_t* qbase = Q  + (size_t)b * T_ * qld  + (size_t)h  * DH_;
  const bf16_t* kbase = Km + (size_t)b * T_ * kvld + (size_t)hk * DH_;
  const bf16_t* vbase = Vm + (size_t)b * T_ * kvld + (size_t)hk * DH_;

  // Q fragments: 4 chunks of K=32 along DH
  v16bf qf[4];
  {
    const bf16_t* qp = qbase + (size_t)(q0 + l15) * qld;
#pragma unroll
    for (int kc = 0; kc < 4; ++kc) {
      const bf16_t* p  = qp + kc * 32 + half * 16;
      v8bf lo = *(const v8bf*)(p);
      v8bf hi = *(const v8bf*)(p + 8);
#pragma unroll
      for (int i = 0; i < 8; ++i) { qf[kc][i] = lo[i]; qf[kc][8 + i] = hi[i]; }
    }
  }

  v8f oacc[8] = {};
  float mrow[8], lrow[8];
#pragma unroll
  for (int r = 0; r < 8; ++r) { mrow[r] = -3.0e38f; lrow[r] = 0.0f; }

  const float sm_scale = 0.08838834764831845f;   // 1/sqrt(DH)

  const int ntiles = (q0 + 16 + 31) / 32;        // keys [0, q0+16) needed

  // prologue: kick off tile 0 DMA
  tdm_load_v_tile(vbase, (unsigned)(uintptr_t)&lds_v[0][0][0]);

  for (int kt = 0; kt < ntiles; ++kt) {
    const int k0  = kt * 32;
    const int buf = kt & 1;

    if (kt + 1 < ntiles) {
      // buffer buf^1 was read last iteration; drain LDS reads, then DMA
      asm volatile("s_wait_dscnt 0x0" ::: "memory");
      tdm_load_v_tile(vbase + (size_t)(kt + 1) * 32 * kvld,
                      (unsigned)(uintptr_t)&lds_v[buf ^ 1][0][0]);
      __builtin_amdgcn_s_wait_tensorcnt(1);   // tile kt resident
    } else {
      __builtin_amdgcn_s_wait_tensorcnt(0);   // last tile resident
    }

    // ---- S = Q K^T (two 16-key fragments), K frags straight from global ----
    v8f sf[2] = {};
#pragma unroll
    for (int nf = 0; nf < 2; ++nf) {
#pragma unroll
      for (int kc = 0; kc < 4; ++kc) {
        const bf16_t* kp =
            kbase + (size_t)(k0 + nf * 16 + l15) * kvld + kc * 32 + half * 16;
        v16bf kb = *(const v16bf*)kp;
        sf[nf] = __builtin_amdgcn_wmma_f32_16x16x32_bf16(
            false, qf[kc], false, kb, (short)0, sf[nf], false, false);
      }
    }

    // ---- scale + causal mask (C layout: col = k0+nf*16+l15, row = q0+r+8h) --
#pragma unroll
    for (int nf = 0; nf < 2; ++nf) {
      const int col = k0 + nf * 16 + l15;
#pragma unroll
      for (int r = 0; r < 8; ++r) {
        const int row = q0 + r + 8 * half;
        const float s = sf[nf][r] * sm_scale;
        sf[nf][r] = (col > row) ? -3.0e38f : s;
      }
    }

    // ---- online softmax: row reductions within 16-lane groups ----
#pragma unroll
    for (int r = 0; r < 8; ++r) {
      float m = fmaxf(sf[0][r], sf[1][r]);
      m = fmaxf(m, __shfl_xor(m, 1, 16));
      m = fmaxf(m, __shfl_xor(m, 2, 16));
      m = fmaxf(m, __shfl_xor(m, 4, 16));
      m = fmaxf(m, __shfl_xor(m, 8, 16));
      const float nm = fmaxf(mrow[r], m);

      const float p0 = __expf(sf[0][r] - nm);
      const float p1 = __expf(sf[1][r] - nm);
      sf[0][r] = p0;
      sf[1][r] = p1;
      float s = p0 + p1;
      s += __shfl_xor(s, 1, 16);
      s += __shfl_xor(s, 2, 16);
      s += __shfl_xor(s, 4, 16);
      s += __shfl_xor(s, 8, 16);

      const float alpha = __expf(mrow[r] - nm);
      lrow[r] = lrow[r] * alpha + s;
      mrow[r] = nm;
#pragma unroll
      for (int f = 0; f < 8; ++f) oacc[f][r] = oacc[f][r] * alpha;
    }

    // ---- relayout P (C layout) -> A fragment via LDS ----
#pragma unroll
    for (int nf = 0; nf < 2; ++nf)
#pragma unroll
      for (int r = 0; r < 8; ++r)
        lds_p[r + 8 * half][nf * 16 + l15] = (bf16_t)sf[nf][r];
    __syncthreads();

    v16bf pa;
    {
      const bf16_t* pp = &lds_p[l15][0];
#pragma unroll
      for (int i = 0; i < 8; ++i) {
        pa[i]     = pp[half * 16 + i];
        pa[8 + i] = pp[half * 16 + 8 + i];
      }
    }

    // ---- O += P V : B frag gathered from row-major TDM tile ----
#pragma unroll
    for (int f = 0; f < 8; ++f) {
      v16bf vb;
#pragma unroll
      for (int e = 0; e < 16; ++e)
        vb[e] = lds_v[buf][16 * half + e][f * 16 + l15];
      oacc[f] = __builtin_amdgcn_wmma_f32_16x16x32_bf16(
          false, pa, false, vb, (short)0, oacc[f], false, false);
    }
  }

  // ---- normalize and store O (bf16) ----
  float inv[8];
#pragma unroll
  for (int r = 0; r < 8; ++r) inv[r] = 1.0f / lrow[r];

  bf16_t* obase = O + (size_t)b * T_ * qld + (size_t)h * DH_;
#pragma unroll
  for (int f = 0; f < 8; ++f) {
    const int dh = f * 16 + l15;
#pragma unroll
    for (int r = 0; r < 8; ++r) {
      const int row = q0 + r + 8 * half;
      obase[(size_t)row * qld + dh] = (bf16_t)(oacc[f][r] * inv[r]);
    }
  }
}

// ---------------------------------------------------------------------------
// Host-side orchestration
// ---------------------------------------------------------------------------
extern "C" void kernel_launch(void* const* d_in, const int* in_sizes, int n_in,
                              void* d_out, int out_size, void* d_ws,
                              size_t ws_size, hipStream_t stream) {
  const float* x    = (const float*)d_in[0];
  const float* cosT = (const float*)d_in[1];
  const float* sinT = (const float*)d_in[2];
  const float* Wq   = (const float*)d_in[3];
  const float* Wk   = (const float*)d_in[4];
  const float* Wv   = (const float*)d_in[5];
  const float* Wo   = (const float*)d_in[6];
  const float* temp = (const float*)d_in[7];
  float* out = (float*)d_out;

  const int NT  = B_ * T_;                // 4096 rows
  const int nX  = NT * D_;
  const int nWq = D_ * D_;
  const int nWk = (HKV_ * DH_) * D_;
  const int nWo = D_ * D_;

  bf16_t* ws = (bf16_t*)d_ws;
  size_t off = 0;
  bf16_t* xb  = ws + off; off += (size_t)nX;
  bf16_t* Wqb = ws + off; off += (size_t)nWq;
  bf16_t* Wkb = ws + off; off += (size_t)nWk;
  bf16_t* Wvb = ws + off; off += (size_t)nWk;
  bf16_t* Wob = ws + off; off += (size_t)nWo;
  bf16_t* Qb  = ws + off; off += (size_t)NT * D_;
  bf16_t* Kb  = ws + off; off += (size_t)NT * (HKV_ * DH_);
  bf16_t* Vb  = ws + off; off += (size_t)NT * (HKV_ * DH_);
  bf16_t* Ob  = ws + off; off += (size_t)NT * D_;

  // 1) casts to bf16
  cast_f32_bf16<<<(nX  + 255) / 256, 256, 0, stream>>>(x,  xb,  nX);
  cast_f32_bf16<<<(nWq + 255) / 256, 256, 0, stream>>>(Wq, Wqb, nWq);
  cast_f32_bf16<<<(nWk + 255) / 256, 256, 0, stream>>>(Wk, Wkb, nWk);
  cast_f32_bf16<<<(nWk + 255) / 256, 256, 0, stream>>>(Wv, Wvb, nWk);
  cast_f32_bf16<<<(nWo + 255) / 256, 256, 0, stream>>>(Wo, Wob, nWo);

  // 2) QKV projections (WMMA GEMMs, bf16 out)
  gemm_bf16_kernel<false><<<dim3(NT / 16, D_ / 128), 32, 0, stream>>>(
      xb, Wqb, nullptr, Qb, NT, D_, D_);
  gemm_bf16_kernel<false><<<dim3(NT / 16, (HKV_ * DH_) / 128), 32, 0, stream>>>(
      xb, Wkb, nullptr, Kb, NT, HKV_ * DH_, D_);
  gemm_bf16_kernel<false><<<dim3(NT / 16, (HKV_ * DH_) / 128), 32, 0, stream>>>(
      xb, Wvb, nullptr, Vb, NT, HKV_ * DH_, D_);

  // 3) RoPE (+temp on q)
  rope_kernel<<<dim3(NT, H_),   DH_, 0, stream>>>(Qb, cosT, sinT, temp, H_, 1);
  rope_kernel<<<dim3(NT, HKV_), DH_, 0, stream>>>(Kb, cosT, sinT, temp, HKV_, 0);

  // 4) causal flash attention (WMMA + TDM double-buffered V)
  attn_kernel<<<dim3(T_ / 16, H_, B_), 32, 0, stream>>>(Qb, Kb, Vb, Ob);

  // 5) output projection (WMMA GEMM, f32 out)
  gemm_bf16_kernel<true><<<dim3(NT / 16, D_ / 128), 32, 0, stream>>>(
      Ob, Wob, out, nullptr, NT, D_, D_);

  (void)in_sizes; (void)n_in; (void)out_size; (void)ws_size;
}
